// GraphTransformerDemon_20143396618415
// MI455X (gfx1250) — matrix operand order repaired and checked by reading, not compiled, not run on hardware
//
#include <hip/hip_runtime.h>
#include <hip/hip_bf16.h>

// ---------------------------------------------------------------------------
// GraphTransformerDemon for MI455X (gfx1250, wave32, WMMA 16x16x32 f16).
//
// Algebraic optimization: the per-pair edge GEMM (eh @ We2) is linear in the
// j-reduction, so we accumulate eh (and keep*eh) over j FIRST and apply We2
// once per node: 589k tiny 64x64 GEMVs become two 1536x64x64 GEMMs.
//
// GEMMs: one block = one 16-row M tile x full N. A tile is DMA'd to LDS via
// global_load_async_to_lds_b128 (ASYNCcnt), W tile staged transposed as f16
// so B fragments are single 32B LDS vector loads. K fully unrolled (template)
// -> back-to-back v_wmma_f32_16x16x32_f16. All fragment builds are
// branch-free (vector loads + statically-indexed selects).
// ---------------------------------------------------------------------------

#define Bsz 4
#define Ssz 384
#define Hsz 64
#define NHh 4
#define DHd 16
#define INs 32

typedef __attribute__((ext_vector_type(16))) _Float16 v16h;
typedef __attribute__((ext_vector_type(8)))  float    v8f;

// A-matrix 16x32 f16 layout (ISA 7.12.2): lane<16 holds K{0..7,16..23},
// lane>=16 holds K{8..15,24..31}, 2 halves per VGPR.
// gemm_t and P@V use contiguous K in A and B: lanes 0-15 hold K=0..15, lanes 16-31 K=16..31.

// Build an A fragment from 32 contiguous f32 at `base` (16B aligned).
// kb = 8*(lane>>4). Reads base[2*kb..2*kb+15] as float4s.
__device__ __forceinline__ v16h a_frag_f32(const float* base, int kb) {
  const float4 x0 = *(const float4*)(base + 2 * kb);
  const float4 x1 = *(const float4*)(base + 2 * kb + 4);
  const float4 x2 = *(const float4*)(base + 2 * kb + 8);
  const float4 x3 = *(const float4*)(base + 2 * kb + 12);
  v16h a;
  a[0]  = (_Float16)x0.x; a[1]  = (_Float16)x0.y;
  a[2]  = (_Float16)x0.z; a[3]  = (_Float16)x0.w;
  a[4]  = (_Float16)x1.x; a[5]  = (_Float16)x1.y;
  a[6]  = (_Float16)x1.z; a[7]  = (_Float16)x1.w;
  a[8]  = (_Float16)x2.x; a[9]  = (_Float16)x2.y;
  a[10] = (_Float16)x2.z; a[11] = (_Float16)x2.w;
  a[12] = (_Float16)x3.x; a[13] = (_Float16)x3.y;
  a[14] = (_Float16)x3.z; a[15] = (_Float16)x3.w;
  return a;
}

// ---------------------------------------------------------------------------
// Templated WMMA GEMM: C[M,N] = act(A[M,K] @ W[K,N] + bias).
// grid = (1, M/16), block = 2*N threads = N/16 waves; wave w owns N-tile w.
// ---------------------------------------------------------------------------
template <int K, int N, bool RELU, bool HASBIAS>
__global__ __launch_bounds__(2 * N) void gemm_t(const float* __restrict__ A,
                                                const float* __restrict__ W,
                                                const float* __restrict__ bias,
                                                float* __restrict__ C) {
  constexpr int THREADS = 2 * N;
  __shared__ __align__(16) float     Asf[16][K];  // raw f32 A tile (async DMA)
  __shared__ __align__(32) _Float16  Ws[N][K];    // W transposed, f16
  const int tid = threadIdx.x;
  const int m0 = blockIdx.y * 16;

  // --- A tile: 16 rows x K = 16*K CONTIGUOUS floats -> async DMA to LDS ---
  {
    const uint32_t lds0 = (uint32_t)(uintptr_t)&Asf[0][0];
    const float* src = A + m0 * K;
#pragma unroll
    for (int i = tid; i < K * 4; i += THREADS) {   // K*4 x 16B transfers
      const uint32_t dst = lds0 + (uint32_t)i * 16u;
      const float* s = src + i * 4;
      asm volatile("global_load_async_to_lds_b128 %0, %1, off"
                   :: "v"(dst), "v"(s) : "memory");
    }
  }
  // --- W tile: convert to f16, store transposed (coalesced global reads) ---
  for (int idx = tid; idx < K * N; idx += THREADS) {
    const int kk = idx / N, nn = idx - kk * N;
    Ws[nn][kk] = (_Float16)W[idx];
  }
  asm volatile("s_wait_asynccnt 0x0" ::: "memory");
  __syncthreads();

  const int wave = tid >> 5, lane = tid & 31;
  const int n0 = wave * 16;
  const int row = lane & 15, hi = lane >> 4;
  const int kb = hi * 8;

  v8f acc = {};
#pragma unroll
  for (int k0 = 0; k0 < K; k0 += 32) {
    const v16h a = a_frag_f32(&Asf[row][k0], kb);       // 4x ds_load_b128
    // 16 contiguous f16 at 32B-aligned offset: 2x ds_load_b128
    const v16h b = *(const v16h*)&Ws[n0 + row][k0 + 16 * hi];
    acc = __builtin_amdgcn_wmma_f32_16x16x32_f16(false, a, false, b,
                                                 (short)0, acc, false, false);
  }

  float bv = 0.f;
  if constexpr (HASBIAS) bv = bias[n0 + row];
  (void)bias;
#pragma unroll
  for (int r = 0; r < 8; ++r) {
    float v = acc[r] + bv;
    if constexpr (RELU) v = fmaxf(v, 0.f);
    C[(m0 + r + 8 * hi) * N + n0 + row] = v;
  }
}

// ---------------------------------------------------------------------------
// Attention: one wave per (b, head, 16-query tile). Scores via WMMA with
// DH=16 zero-padded to K=32 (branch-free: unconditional b128 loads + static
// cndmask selects); f32 softmax over an LDS [16][388] row tile; P@V via 12
// WMMA k-steps reading P from LDS with vectorized loads.
// grid = (S/16, NH, B), block = 32.
// ---------------------------------------------------------------------------
__global__ void attn_kernel(const float* __restrict__ q,
                            const float* __restrict__ k,
                            const float* __restrict__ v,
                            float* __restrict__ att) {
  __shared__ float P[16][388];  // +4 pad vs 64-bank LDS
  const int lane = threadIdx.x;
  const int qt = blockIdx.x, head = blockIdx.y, b = blockIdx.z;
  const int row = lane & 15, hi = lane >> 4;
  const int kb = hi * 8;
  const int hbase = head * DHd;
  const bool lo = (hi == 0);

  // --- scores = (Q @ K^T) * 1/sqrt(DH) ---
  // A frag (K padded 16->32): live halves t=0..7 hold dims kb..kb+7.
  v16h a;
  {
    const float* qrow = q + (b * Ssz + qt * 16 + row) * Hsz + hbase;
    const float4 q0 = *(const float4*)(qrow);
    const float4 q1 = *(const float4*)(qrow + 4);
    const float4 q2 = *(const float4*)(qrow + 8);
    const float4 q3 = *(const float4*)(qrow + 12);
    a[0] = (_Float16)(lo ? q0.x : q2.x); a[1] = (_Float16)(lo ? q0.y : q2.y);
    a[2] = (_Float16)(lo ? q0.z : q2.z); a[3] = (_Float16)(lo ? q0.w : q2.w);
    a[4] = (_Float16)(lo ? q1.x : q3.x); a[5] = (_Float16)(lo ? q1.y : q3.y);
    a[6] = (_Float16)(lo ? q1.z : q3.z); a[7] = (_Float16)(lo ? q1.w : q3.w);
#pragma unroll
    for (int t = 8; t < 16; ++t) a[t] = (_Float16)0.f;  // padded K
  }
#pragma unroll 4
  for (int jt = 0; jt < Ssz / 16; ++jt) {
    // B frag: halves t=0..7 hold dims kb..kb+7, t=8..15 padded 0.
    const float* krow = k + (b * Ssz + jt * 16 + row) * Hsz + hbase;
    const float4 k0v = *(const float4*)(krow);
    const float4 k1v = *(const float4*)(krow + 4);
    const float4 k2v = *(const float4*)(krow + 8);
    const float4 k3v = *(const float4*)(krow + 12);
    const float z = 0.f;
    v16h bm;
    bm[0]  = (_Float16)(lo ? k0v.x : k2v.x); bm[1]  = (_Float16)(lo ? k0v.y : k2v.y);
    bm[2]  = (_Float16)(lo ? k0v.z : k2v.z); bm[3]  = (_Float16)(lo ? k0v.w : k2v.w);
    bm[4]  = (_Float16)(lo ? k1v.x : k3v.x); bm[5]  = (_Float16)(lo ? k1v.y : k3v.y);
    bm[6]  = (_Float16)(lo ? k1v.z : k3v.z); bm[7]  = (_Float16)(lo ? k1v.w : k3v.w);
    bm[8]  = (_Float16)z; bm[9]  = (_Float16)z;
    bm[10] = (_Float16)z; bm[11] = (_Float16)z;
    bm[12] = (_Float16)z; bm[13] = (_Float16)z;
    bm[14] = (_Float16)z; bm[15] = (_Float16)z;
    v8f c = {};
    c = __builtin_amdgcn_wmma_f32_16x16x32_f16(false, a, false, bm,
                                               (short)0, c, false, false);
#pragma unroll
    for (int r = 0; r < 8; ++r)
      P[r + 8 * hi][jt * 16 + row] = c[r] * 0.25f;  // 1/sqrt(16)
  }
  __syncthreads();

  // --- row softmax (wave32 shuffle reductions) ---
  for (int r = 0; r < 16; ++r) {
    float m = -1e30f;
    for (int c0 = lane; c0 < Ssz; c0 += 32) m = fmaxf(m, P[r][c0]);
#pragma unroll
    for (int o = 16; o > 0; o >>= 1) m = fmaxf(m, __shfl_xor(m, o, 32));
    float s = 0.f;
    for (int c0 = lane; c0 < Ssz; c0 += 32) {
      const float e = __expf(P[r][c0] - m);
      P[r][c0] = e;
      s += e;
    }
#pragma unroll
    for (int o = 16; o > 0; o >>= 1) s += __shfl_xor(s, o, 32);
    const float inv = 1.f / s;
    for (int c0 = lane; c0 < Ssz; c0 += 32) P[r][c0] *= inv;
  }
  __syncthreads();

  // --- att = P @ V ---
  v8f acc = {};
#pragma unroll 3
  for (int jt = 0; jt < Ssz / 32; ++jt) {
    const v16h ap = a_frag_f32(&P[row][jt * 32], kb);   // 4x ds_load_b128
    v16h bv;
#pragma unroll
    for (int t = 0; t < 16; ++t)
      bv[t] = (_Float16)v[(b * Ssz + jt * 32 + 16 * hi + t) * Hsz + hbase + row];
    acc = __builtin_amdgcn_wmma_f32_16x16x32_f16(false, ap, false, bv,
                                                 (short)0, acc, false, false);
  }
#pragma unroll
  for (int r = 0; r < 8; ++r)
    att[(b * Ssz + qt * 16 + r + 8 * hi) * Hsz + hbase + row] = acc[r];
}

// ---------------------------------------------------------------------------
// out = LayerNorm(A + Bv) * g + beta. One wave per row; lane handles h,h+32.
// grid = B*S, block = 32.
// ---------------------------------------------------------------------------
__global__ void add_ln_kernel(const float* __restrict__ A,
                              const float* __restrict__ Bv,
                              const float* __restrict__ g,
                              const float* __restrict__ beta,
                              float* __restrict__ out) {
  const int row = blockIdx.x, l = threadIdx.x;
  const float s0 = A[row * Hsz + l]      + Bv[row * Hsz + l];
  const float s1 = A[row * Hsz + l + 32] + Bv[row * Hsz + l + 32];
  float s = s0 + s1;
#pragma unroll
  for (int o = 16; o > 0; o >>= 1) s += __shfl_xor(s, o, 32);
  const float mean = s * (1.f / Hsz);
  const float d0 = s0 - mean, d1 = s1 - mean;
  float vv = d0 * d0 + d1 * d1;
#pragma unroll
  for (int o = 16; o > 0; o >>= 1) vv += __shfl_xor(vv, o, 32);
  const float rstd = rsqrtf(vv * (1.f / Hsz) + 1e-5f);
  out[row * Hsz + l]      = d0 * rstd * g[l]      + beta[l];
  out[row * Hsz + l + 32] = d1 * rstd * g[l + 32] + beta[l + 32];
}

// ---------------------------------------------------------------------------
// keep[b,i,j] = sigmoid(relu(Ld[i]+Rd[j]) . Wd2 + bd2). (bd1 folded into Rd.)
// One thread per pair.
// ---------------------------------------------------------------------------
__global__ void keep_kernel(const float* __restrict__ Ld,
                            const float* __restrict__ Rd,
                            const float* __restrict__ Wd2,
                            const float* __restrict__ bd2,
                            float* __restrict__ keep) {
  const int pair = blockIdx.x * blockDim.x + threadIdx.x;
  if (pair >= Bsz * Ssz * Ssz) return;
  const int b = pair / (Ssz * Ssz);
  const int rem = pair - b * Ssz * Ssz;
  const int i = rem / Ssz, j = rem - i * Ssz;
  const float* ld = Ld + (b * Ssz + i) * 32;
  const float* rd = Rd + (b * Ssz + j) * 32;
  float acc = bd2[0];
#pragma unroll
  for (int d = 0; d < 32; ++d)
    acc += fmaxf(ld[d] + rd[d], 0.f) * Wd2[d];
  keep[pair] = 1.f / (1.f + __expf(-acc));
}

// ---------------------------------------------------------------------------
// For each node i: totpre[i,h] = sum_j relu(Lh[i,h]+Rh[j,h]),
//                  actpre[i,h] = sum_j keep[i,j]*relu(...),  ksum[i]=sum_j keep.
// (be1 folded into Rh.) grid = B*S, block = 64 (thread = channel h).
// ---------------------------------------------------------------------------
__global__ void pair_acc_kernel(const float* __restrict__ Lh,
                                const float* __restrict__ Rh,
                                const float* __restrict__ keep,
                                float* __restrict__ actpre,
                                float* __restrict__ totpre,
                                float* __restrict__ ksum) {
  const int bi = blockIdx.x;       // b*S+i
  const int b = bi / Ssz;
  const int i = bi - b * Ssz;
  const int h = threadIdx.x;
  const float lv = Lh[bi * Hsz + h];
  const float* kr = keep + (b * Ssz + i) * Ssz;
  const float* rr = Rh + b * Ssz * Hsz + h;
  float at = 0.f, aa = 0.f, ks = 0.f;
  for (int j = 0; j < Ssz; ++j) {
    const float kp = kr[j];
    const float e = fmaxf(lv + rr[j * Hsz], 0.f);
    at += e;
    aa += kp * e;
    ks += kp;
  }
  totpre[bi * Hsz + h] = at;
  actpre[bi * Hsz + h] = aa;
  if (h == 0) ksum[bi] = ks;
}

// ---------------------------------------------------------------------------
// Pool + classifier. U = actpre@We2, T = totpre@We2 (be2 terms added here).
// grid = B, block = 64.
// ---------------------------------------------------------------------------
__global__ void pool_cls_kernel(const float* __restrict__ nodes,
                                const float* __restrict__ U,
                                const float* __restrict__ T,
                                const float* __restrict__ ksum,
                                const float* __restrict__ be2,
                                const float* __restrict__ Wc1,
                                const float* __restrict__ bc1,
                                const float* __restrict__ Wc2,
                                const float* __restrict__ bc2,
                                float* __restrict__ out) {
  __shared__ float cat[2 * Hsz];
  __shared__ float hc[32];
  const int b = blockIdx.x, t = threadIdx.x;  // t = channel h (0..63)
  float pa = 0.f, pt = 0.f, ka = 0.f;
  for (int i = 0; i < Ssz; ++i) {
    const int idx = (b * Ssz + i) * Hsz + t;
    const float u = U[idx];
    pa += nodes[idx] + u;
    pt += T[idx] - u;
    ka += ksum[b * Ssz + i];
  }
  const float be = be2[t];
  pa = (pa + be * ka) * (1.f / Ssz);
  pt = (pt + be * ((float)Ssz * (float)Ssz - ka)) * (1.f / Ssz);
  cat[t] = pa;
  cat[Hsz + t] = pt;
  __syncthreads();
  if (t < 32) {
    float acc = bc1[t];
    for (int d = 0; d < 2 * Hsz; ++d) acc += cat[d] * Wc1[d * 32 + t];
    hc[t] = fmaxf(acc, 0.f);
  }
  __syncthreads();
  if (t < 3) {
    float acc = bc2[t];
    for (int d = 0; d < 32; ++d) acc += hc[d] * Wc2[d * 3 + t];
    out[b * 3 + t] = acc;
  }
}

// ---------------------------------------------------------------------------
// Launch
// ---------------------------------------------------------------------------
extern "C" void kernel_launch(void* const* d_in, const int* in_sizes, int n_in,
                              void* d_out, int out_size, void* d_ws, size_t ws_size,
                              hipStream_t stream) {
  (void)in_sizes; (void)n_in; (void)out_size; (void)ws_size;
  const float* x   = (const float*)d_in[0];
  const float* Wp  = (const float*)d_in[1];  const float* bp  = (const float*)d_in[2];
  const float* Wq  = (const float*)d_in[3];  const float* bq  = (const float*)d_in[4];
  const float* Wk  = (const float*)d_in[5];  const float* bk  = (const float*)d_in[6];
  const float* Wv  = (const float*)d_in[7];  const float* bv  = (const float*)d_in[8];
  const float* Wo  = (const float*)d_in[9];  const float* bo  = (const float*)d_in[10];
  const float* g1  = (const float*)d_in[11]; const float* b1  = (const float*)d_in[12];
  const float* Wf1 = (const float*)d_in[13]; const float* bf1 = (const float*)d_in[14];
  const float* Wf2 = (const float*)d_in[15]; const float* bf2 = (const float*)d_in[16];
  const float* g2  = (const float*)d_in[17]; const float* b2  = (const float*)d_in[18];
  const float* We1 = (const float*)d_in[19]; const float* be1 = (const float*)d_in[20];
  const float* We2 = (const float*)d_in[21]; const float* be2 = (const float*)d_in[22];
  const float* Wd1 = (const float*)d_in[23]; const float* bd1 = (const float*)d_in[24];
  const float* Wd2 = (const float*)d_in[25]; const float* bd2 = (const float*)d_in[26];
  const float* Wc1 = (const float*)d_in[27]; const float* bc1 = (const float*)d_in[28];
  const float* Wc2 = (const float*)d_in[29]; const float* bc2 = (const float*)d_in[30];
  float* out = (float*)d_out;

  const int M = Bsz * Ssz;                 // 1536
  float* ws = (float*)d_ws;
  // workspace layout (float offsets)
  float* n0     = ws;                      // 98304
  float* q      = ws + 98304;
  float* k      = ws + 196608;
  float* v      = ws + 294912;
  float* att    = ws + 393216;
  float* o      = ws + 491520;
  float* n1     = ws + 589824;
  float* f      = ws + 688128;             // 196608 (N=128)
  float* ff     = ws + 884736;
  float* nodes  = ws + 983040;
  float* Lh     = ws + 1081344;
  float* Rh     = ws + 1179648;
  float* Ld     = ws + 1277952;            // 49152 (N=32)
  float* Rd     = ws + 1327104;
  float* keep   = ws + 1376256;            // 589824
  float* ksum   = ws + 1966080;            // 1536
  float* actpre = ws + 1967616;
  float* totpre = ws + 2065920;
  float* U      = ws + 2164224;
  float* T      = ws + 2262528;            // end = 2360832 floats (~9.0 MB)

  const dim3 gM(1, M / 16);                // 96 M-tiles

  // 1. projector: n0 = relu(x @ Wp + bp)      [1536x32]@[32x64]
  gemm_t<INs, Hsz, true,  true ><<<gM, 128, 0, stream>>>(x, Wp, bp, n0);
  // 2. q/k/v
  gemm_t<Hsz, Hsz, false, true ><<<gM, 128, 0, stream>>>(n0, Wq, bq, q);
  gemm_t<Hsz, Hsz, false, true ><<<gM, 128, 0, stream>>>(n0, Wk, bk, k);
  gemm_t<Hsz, Hsz, false, true ><<<gM, 128, 0, stream>>>(n0, Wv, bv, v);
  // 3. attention
  attn_kernel<<<dim3(Ssz / 16, NHh, Bsz), 32, 0, stream>>>(q, k, v, att);
  // 4. output projection + residual LN
  gemm_t<Hsz, Hsz, false, true ><<<gM, 128, 0, stream>>>(att, Wo, bo, o);
  add_ln_kernel<<<M, 32, 0, stream>>>(n0, o, g1, b1, n1);
  // 5. FFN + residual LN
  gemm_t<Hsz, 128, true,  true ><<<gM, 256, 0, stream>>>(n1, Wf1, bf1, f);
  gemm_t<128, Hsz, false, true ><<<gM, 128, 0, stream>>>(f, Wf2, bf2, ff);
  add_ln_kernel<<<M, 32, 0, stream>>>(n1, ff, g2, b2, nodes);
  // 6. edge / demon projections (be1, bd1 folded into right-hand halves)
  gemm_t<Hsz, Hsz, false, false><<<gM, 128, 0, stream>>>(nodes, We1, nullptr, Lh);
  gemm_t<Hsz, Hsz, false, true ><<<gM, 128, 0, stream>>>(nodes, We1 + Hsz * Hsz, be1, Rh);
  gemm_t<Hsz, 32,  false, false><<<gM, 64, 0, stream>>>(nodes, Wd1, nullptr, Ld);
  gemm_t<Hsz, 32,  false, true ><<<gM, 64, 0, stream>>>(nodes, Wd1 + Hsz * 32, bd1, Rd);
  // 7. demon gates + pair accumulation (j-reduction BEFORE We2)
  keep_kernel<<<(Bsz * Ssz * Ssz + 255) / 256, 256, 0, stream>>>(Ld, Rd, Wd2, bd2, keep);
  pair_acc_kernel<<<M, 64, 0, stream>>>(Lh, Rh, keep, actpre, totpre, ksum);
  // 8. apply We2 once per node
  gemm_t<Hsz, Hsz, false, false><<<gM, 128, 0, stream>>>(actpre, We2, nullptr, U);
  gemm_t<Hsz, Hsz, false, false><<<gM, 128, 0, stream>>>(totpre, We2, nullptr, T);
  // 9. pooling + classifier
  pool_cls_kernel<<<Bsz, 64, 0, stream>>>(nodes, U, T, ksum, be2, Wc1, bc1,
                                          Wc2, bc2, out);
}
